// LeukemiaGNN_74036646248622
// MI455X (gfx1250) — matrix operand order repaired and checked by reading, not compiled, not run on hardware
//
#include <hip/hip_runtime.h>
#include <hip/hip_bf16.h>

typedef __attribute__((ext_vector_type(16))) __bf16 v16bf;
typedef __attribute__((ext_vector_type(8)))  float  v8f;

#define NNODES 12800
#define NEDGES 204800
#define NGRAPH 128
#define GFEAT  12800   // 128 * 100

// ---------------------------------------------------------------------------
// Utility: zero a float buffer (n4 = count/4)
// ---------------------------------------------------------------------------
__global__ void zero_f32(float* __restrict__ p, unsigned n4) {
    unsigned i = blockIdx.x * blockDim.x + threadIdx.x;
    if (i < n4) {
        float4 z; z.x = 0.f; z.y = 0.f; z.z = 0.f; z.w = 0.f;
        reinterpret_cast<float4*>(p)[i] = z;
    }
}

// ---------------------------------------------------------------------------
// deg[row[e]] += ew[e]
// ---------------------------------------------------------------------------
__global__ void deg_kernel(const int* __restrict__ row,
                           const float* __restrict__ ew,
                           float* __restrict__ deg) {
    unsigned e = blockIdx.x * blockDim.x + threadIdx.x;
    if (e < NEDGES) atomicAdd(deg + row[e], ew[e]);
}

// deg -> dinv in place
__global__ void dinv_kernel(float* __restrict__ deg) {
    unsigned i = blockIdx.x * blockDim.x + threadIdx.x;
    if (i < NNODES) {
        float d = deg[i];
        deg[i] = (d > 0.f) ? rsqrtf(fmaxf(d, 1e-30f)) : 0.f;
    }
}

// norm[e] = -(dinv[row] * ew * dinv[col])
__global__ void norm_kernel(const int* __restrict__ row,
                            const int* __restrict__ col,
                            const float* __restrict__ ew,
                            const float* __restrict__ dinv,
                            float* __restrict__ norm) {
    unsigned e = blockIdx.x * blockDim.x + threadIdx.x;
    if (e < NEDGES) norm[e] = -(dinv[row[e]] * ew[e] * dinv[col[e]]);
}

// ---------------------------------------------------------------------------
// SpMM propagate: out[row[e], :] += norm[e] * t[col[e], :]
// One thread handles 4 consecutive features of one edge.
// ---------------------------------------------------------------------------
__global__ void spmm_prop(const int* __restrict__ row,
                          const int* __restrict__ col,
                          const float* __restrict__ norm,
                          const float* __restrict__ t,
                          float* __restrict__ out,
                          int F, unsigned nchunks, unsigned total) {
    unsigned id = blockIdx.x * blockDim.x + threadIdx.x;
    if (id >= total) return;
    unsigned e  = id / nchunks;
    unsigned f4 = (id - e * nchunks) * 4u;
    float w = norm[e];
    int c = col[e], r = row[e];
    const float4 v = *reinterpret_cast<const float4*>(t + (size_t)c * F + f4);
    float* o = out + (size_t)r * F + f4;
    atomicAdd(o + 0, w * v.x);
    atomicAdd(o + 1, w * v.y);
    atomicAdd(o + 2, w * v.z);
    atomicAdd(o + 3, w * v.w);
}

// ---------------------------------------------------------------------------
// Weight transpose + bf16 convert:  Wt[p][n][k] = (bf16) W[p][k][n]
// Layout (3, fo, fi) with k fastest => a lane's B fragment is 16 contiguous
// bf16 (32 bytes): two global_load_b128 per fragment.
// ---------------------------------------------------------------------------
__global__ void wt_transpose_bf16(const float* __restrict__ W,
                                  __bf16* __restrict__ Wt,
                                  int fi, int fo, unsigned total) {
    unsigned i = blockIdx.x * blockDim.x + threadIdx.x;
    if (i >= total) return;
    unsigned pf  = (unsigned)(fi * fo);
    unsigned p   = i / pf;
    unsigned rem = i - p * pf;
    unsigned n   = rem / (unsigned)fi;
    unsigned k   = rem - n * (unsigned)fi;
    Wt[i] = (__bf16)W[(size_t)p * pf + (size_t)k * fo + n];
}

// ---------------------------------------------------------------------------
// WMMA helpers
// ---------------------------------------------------------------------------
__device__ __forceinline__ v8f wmma_bf16(v16bf a, v16bf b, v8f c) {
    return __builtin_amdgcn_wmma_f32_16x16x32_bf16(
        false, a, false, b, (short)0, c, false, false);
}

// A fragment: contiguous K order (matches B).  Lane (l16, half):
//   element s : K = kb + 16*half + s     -> contiguous run [kb+16h, kb+16h+15]
// PHASE==2 folds Tx2 = 2*P2 - Tx0 into the fragment.
template <int PHASE>
__device__ __forceinline__ v16bf load_a_frag(const float* __restrict__ aRow,
                                             const float* __restrict__ t0Row,
                                             int kb, int half) {
    const float* p0 = aRow + kb + 16 * half;
    const float* p1 = aRow + kb + 16 * half + 8;
    float4 q0 = *reinterpret_cast<const float4*>(p0);
    float4 q1 = *reinterpret_cast<const float4*>(p0 + 4);
    float4 q2 = *reinterpret_cast<const float4*>(p1);
    float4 q3 = *reinterpret_cast<const float4*>(p1 + 4);
    if constexpr (PHASE == 2) {
        const float* r0 = t0Row + kb + 16 * half;
        const float* r1 = t0Row + kb + 16 * half + 8;
        float4 s0 = *reinterpret_cast<const float4*>(r0);
        float4 s1 = *reinterpret_cast<const float4*>(r0 + 4);
        float4 s2 = *reinterpret_cast<const float4*>(r1);
        float4 s3 = *reinterpret_cast<const float4*>(r1 + 4);
        q0.x = 2.f * q0.x - s0.x; q0.y = 2.f * q0.y - s0.y;
        q0.z = 2.f * q0.z - s0.z; q0.w = 2.f * q0.w - s0.w;
        q1.x = 2.f * q1.x - s1.x; q1.y = 2.f * q1.y - s1.y;
        q1.z = 2.f * q1.z - s1.z; q1.w = 2.f * q1.w - s1.w;
        q2.x = 2.f * q2.x - s2.x; q2.y = 2.f * q2.y - s2.y;
        q2.z = 2.f * q2.z - s2.z; q2.w = 2.f * q2.w - s2.w;
        q3.x = 2.f * q3.x - s3.x; q3.y = 2.f * q3.y - s3.y;
        q3.z = 2.f * q3.z - s3.z; q3.w = 2.f * q3.w - s3.w;
    }
    v16bf a;
    a[0]  = (__bf16)q0.x; a[1]  = (__bf16)q0.y;
    a[2]  = (__bf16)q0.z; a[3]  = (__bf16)q0.w;
    a[4]  = (__bf16)q1.x; a[5]  = (__bf16)q1.y;
    a[6]  = (__bf16)q1.z; a[7]  = (__bf16)q1.w;
    a[8]  = (__bf16)q2.x; a[9]  = (__bf16)q2.y;
    a[10] = (__bf16)q2.z; a[11] = (__bf16)q2.w;
    a[12] = (__bf16)q3.x; a[13] = (__bf16)q3.y;
    a[14] = (__bf16)q3.z; a[15] = (__bf16)q3.w;
    return a;
}

// One Chebyshev phase: c[2][4] += A_phase(32xfi) @ W_phase(fi x 64)
template <int PHASE>
__device__ __forceinline__ void cheb_phase(const float* __restrict__ A,
                                           const float* __restrict__ T0,
                                           const __bf16* __restrict__ WtP,
                                           int fi, int m0, int n0,
                                           int l16, int half,
                                           v8f (&c)[2][4]) {
    const float* aRow0  = A  + (size_t)(m0 + l16) * fi;
    const float* aRow1  = aRow0 + (size_t)16 * fi;
    const float* t0Row0 = T0 + (size_t)(m0 + l16) * fi;
    const float* t0Row1 = t0Row0 + (size_t)16 * fi;

    for (int kb = 0; kb < fi; kb += 32) {
        v16bf a0 = load_a_frag<PHASE>(aRow0, t0Row0, kb, half);
        v16bf a1 = load_a_frag<PHASE>(aRow1, t0Row1, kb, half);
#pragma unroll
        for (int t = 0; t < 4; ++t) {
            const int n = n0 + t * 16 + l16;
            // B fragment: lane covers K = kb + half*16 .. +15, contiguous
            v16bf b = *reinterpret_cast<const v16bf*>(
                WtP + (size_t)n * fi + kb + half * 16);
            c[0][t] = wmma_bf16(a0, b, c[0][t]);
            c[1][t] = wmma_bf16(a1, b, c[1][t]);
        }
    }
}

// ---------------------------------------------------------------------------
// Fused Chebyshev GEMM:
//   out = relu( Tx0 @ W0 + Tx1 @ W1 + (2*P2 - Tx0) @ W2 + bias )
// One wave per 32(M) x 64(N) output tile; 8 WMMAs per 32-deep k-step.
// ---------------------------------------------------------------------------
__global__ __launch_bounds__(32)
void cheb_gemm_wmma(const float* __restrict__ Tx0,
                    const float* __restrict__ Tx1,
                    const float* __restrict__ P2,
                    const __bf16* __restrict__ Wt,   // (3, fo, fi) bf16
                    const float* __restrict__ bias,  // (fo)
                    float* __restrict__ out,         // (NNODES, fo)
                    int fi, int fo) {
    const int lane = threadIdx.x;
    const int m0   = blockIdx.x * 32;
    const int n0   = blockIdx.y * 64;
    const int l16  = lane & 15;
    const int half = lane >> 4;

    v8f c[2][4] = {};

    const size_t wstride = (size_t)fo * fi;   // bf16 elements per phase
    cheb_phase<0>(Tx0, Tx0, Wt,               fi, m0, n0, l16, half, c);
    cheb_phase<1>(Tx1, Tx0, Wt + wstride,     fi, m0, n0, l16, half, c);
    cheb_phase<2>(P2,  Tx0, Wt + 2 * wstride, fi, m0, n0, l16, half, c);

    // store: C/D layout — lane col n = l16, VGPR v -> row m = v + 8*half
#pragma unroll
    for (int t = 0; t < 4; ++t) {
        const int n = n0 + t * 16 + l16;
        const float bv = bias[n];
#pragma unroll
        for (int f = 0; f < 2; ++f) {
#pragma unroll
            for (int v = 0; v < 8; ++v) {
                const int m = m0 + f * 16 + v + 8 * half;
                float val = c[f][t][v] + bv;
                out[(size_t)m * fo + n] = val > 0.f ? val : 0.f;
            }
        }
    }
}

// ---------------------------------------------------------------------------
// Final FC (2 classes over 12800 features per graph) + softmax
// ---------------------------------------------------------------------------
__global__ __launch_bounds__(256)
void fc_softmax(const float* __restrict__ h,
                const float* __restrict__ fcw,
                const float* __restrict__ fcb,
                float* __restrict__ out) {
    __shared__ float s0[256];
    __shared__ float s1[256];
    const int g = blockIdx.x;
    const float* hg = h + (size_t)g * GFEAT;
    float a0 = 0.f, a1 = 0.f;
    for (int j = threadIdx.x; j < GFEAT; j += 256) {
        float v = hg[j];
        a0 += v * fcw[j];
        a1 += v * fcw[GFEAT + j];
    }
    s0[threadIdx.x] = a0;
    s1[threadIdx.x] = a1;
    __syncthreads();
    for (int s = 128; s > 0; s >>= 1) {
        if (threadIdx.x < (unsigned)s) {
            s0[threadIdx.x] += s0[threadIdx.x + s];
            s1[threadIdx.x] += s1[threadIdx.x + s];
        }
        __syncthreads();
    }
    if (threadIdx.x == 0) {
        float l0 = s0[0] + fcb[0];
        float l1 = s1[0] + fcb[1];
        float mx = fmaxf(l0, l1);
        float e0 = __expf(l0 - mx);
        float e1 = __expf(l1 - mx);
        float inv = 1.f / (e0 + e1);
        out[g * 2 + 0] = e0 * inv;
        out[g * 2 + 1] = e1 * inv;
    }
}

// ---------------------------------------------------------------------------
// Host-side orchestration
// ---------------------------------------------------------------------------
extern "C" void kernel_launch(void* const* d_in, const int* in_sizes, int n_in,
                              void* d_out, int out_size, void* d_ws, size_t ws_size,
                              hipStream_t stream) {
    (void)in_sizes; (void)n_in; (void)out_size; (void)ws_size;

    const float* x   = (const float*)d_in[0];
    const int*   ei  = (const int*)d_in[1];
    const float* ew  = (const float*)d_in[2];
    const float* Wl[6]  = {(const float*)d_in[3],  (const float*)d_in[5],
                           (const float*)d_in[7],  (const float*)d_in[9],
                           (const float*)d_in[11], (const float*)d_in[13]};
    const float* bl[6]  = {(const float*)d_in[4],  (const float*)d_in[6],
                           (const float*)d_in[8],  (const float*)d_in[10],
                           (const float*)d_in[12], (const float*)d_in[14]};
    const float* fcw = (const float*)d_in[15];
    const float* fcb = (const float*)d_in[16];

    const int* row = ei;
    const int* col = ei + NEDGES;

    // workspace layout
    float* deg   = (float*)d_ws;                       // NNODES
    float* norm  = deg   + NNODES;                     // NEDGES
    float* bufT1 = norm  + NEDGES;                     // NNODES*512
    float* bufT2 = bufT1 + (size_t)NNODES * 512;       // NNODES*512
    float* bufA  = bufT2 + (size_t)NNODES * 512;       // NNODES*512
    float* bufB  = bufA  + (size_t)NNODES * 512;       // NNODES*512
    __bf16* bufWt = (__bf16*)(bufB + (size_t)NNODES * 512); // up to 3*512*512 bf16

    // ---- edge preprocessing ----
    zero_f32<<<(NNODES / 4 + 255) / 256, 256, 0, stream>>>(deg, NNODES / 4);
    deg_kernel<<<(NEDGES + 255) / 256, 256, 0, stream>>>(row, ew, deg);
    dinv_kernel<<<(NNODES + 255) / 256, 256, 0, stream>>>(deg);
    norm_kernel<<<(NEDGES + 255) / 256, 256, 0, stream>>>(row, col, ew, deg, norm);

    // ---- 6 ChebConv layers ----
    const int dims_in[6]  = {128, 64, 128, 256, 512, 256};
    const int dims_out[6] = {64, 128, 256, 512, 256, 128};
    const float* cur = x;
    float* nxt = bufA;

    for (int l = 0; l < 6; ++l) {
        const int fi = dims_in[l];
        const int fo = dims_out[l];
        const unsigned nElem4  = (unsigned)(NNODES * fi / 4);
        const unsigned nchunks = (unsigned)(fi / 4);
        const unsigned total   = (unsigned)NEDGES * nchunks;
        const unsigned wtot    = (unsigned)(3 * fi * fo);

        zero_f32<<<(nElem4 + 255) / 256, 256, 0, stream>>>(bufT1, nElem4);
        zero_f32<<<(nElem4 + 255) / 256, 256, 0, stream>>>(bufT2, nElem4);

        // weights -> transposed bf16 (3, fo, fi)
        wt_transpose_bf16<<<(wtot + 255) / 256, 256, 0, stream>>>(
            Wl[l], bufWt, fi, fo, wtot);

        // Tx1 = L_hat @ Tx0
        spmm_prop<<<(total + 255) / 256, 256, 0, stream>>>(
            row, col, norm, cur, bufT1, fi, nchunks, total);
        // P2 = L_hat @ Tx1   (Tx2 = 2*P2 - Tx0 folded into GEMM)
        spmm_prop<<<(total + 255) / 256, 256, 0, stream>>>(
            row, col, norm, bufT1, bufT2, fi, nchunks, total);

        dim3 grid(NNODES / 32, fo / 64, 1);
        cheb_gemm_wmma<<<grid, 32, 0, stream>>>(
            cur, bufT1, bufT2, bufWt, bl[l], nxt, fi, fo);

        cur = nxt;
        nxt = (nxt == bufA) ? bufB : bufA;
    }

    // ---- FC + softmax ----
    fc_softmax<<<NGRAPH, 256, 0, stream>>>(cur, fcw, fcb, (float*)d_out);
}
